// BiLSTM_54056458387816
// MI455X (gfx1250) — matrix operand order repaired and checked by reading, not compiled, not run on hardware
//
#include <hip/hip_runtime.h>
#include <hip/hip_bf16.h>

// ---------------------------------------------------------------------------
// BiLSTM backward-direction final state, B=4096 T=2048 H=32, + MLP head.
//
// 2 cooperating wave32s per 16-batch group (512 waves total). Wave p owns the
// hidden half 16p..16p+15: gate tiles {p, p+2, p+4, p+6} of
//   gates^T (128 x 16) = W_hh (128x32) @ h^T (32x16) + (x_t * w_ih + b)
// Each tile: V_WMMA_F32_16X16X4_F32 (rank-1 x/bias) feeding the C operand of
// V_WMMA_F32_16X16X32_F16 (A = W_hh tile, B = h in f16). D layout =
// (gate rows in VGPRs, batch in lanes): activations / cell update / length
// masking are lane-local. The per-step h exchange goes through LDS
// (double-buffered, one s_barrier per step); the LDS addressing is chosen so
// each wave reads its full f16 B-matrix operand with two contiguous
// ds_load_b128's - no cross-lane shuffles or selects at all.
//
// Sigmoid trick: i/f/o tiles pre-scaled by 0.5 in W_hh/w_ih/bias, so
// sigmoid(z) = fma(0.5, tanh(d), 0.5) with d = 0.5 z.
// ---------------------------------------------------------------------------

typedef __attribute__((ext_vector_type(16))) _Float16     v16h;
typedef __attribute__((ext_vector_type(8)))  float        v8f;
typedef __attribute__((ext_vector_type(2)))  float        v2f;
typedef __attribute__((ext_vector_type(8)))  unsigned int v8u;

#define B_SZ 4096
#define T_SZ 2048
#define H_SZ 32

__device__ __forceinline__ float tanh_f(float x) {
#if __has_builtin(__builtin_amdgcn_tanhf)
    return __builtin_amdgcn_tanhf(x);
#else
    float y;
    asm volatile("v_tanh_f32 %0, %1" : "=v"(y) : "v"(x));
    return y;
#endif
}

__global__ __launch_bounds__(64)
void lstm_bwd_kernel(const float* __restrict__ x, const int* __restrict__ lengths,
                     const float* __restrict__ w_ih, const float* __restrict__ w_hh,
                     const float* __restrict__ b_ih, const float* __restrict__ b_hh,
                     float* __restrict__ h_ws)
{
    // [2 buffers][2 waves][32 lanes] packed-f16 h tiles, uint4 per lane.
    __shared__ uint4 lds4[128];

    const int  lane = threadIdx.x & 31;
    const int  p    = threadIdx.x >> 5;            // wave in pair: hidden half
    const int  base = blockIdx.x * 16;             // 16 batches per block
    const int  m    = lane & 15;                   // batch column within tile
    const bool lo   = lane < 16;
    const int  ph   = lo ? 0 : 1;                  // lane-half phase

    // --- preload this wave's 4 W_hh tiles as f16 A-matrices (16x32) ---
    // tiles {p, p+2, p+4, p+6} = i/f/g/o rows for hidden 16p..16p+15.
    // A layout: lanes 0-15: v0..7 = K0..15;
    //           lanes16-31: v0..7 = K16..31.
    v16h W[4];
    v2f  A2[4];
#pragma unroll
    for (int j = 0; j < 4; ++j) {                  // j: 0=i 1=f 2=g 3=o
        const int   tg = p + 2 * j;
        const float sc = (j == 2) ? 1.0f : 0.5f;   // g-gate unscaled
#pragma unroll
        for (int e = 0; e < 16; ++e) {
            const int K = 16 * ph + e;
            W[j][e] = (_Float16)(sc * w_hh[(16 * tg + m) * H_SZ + K]);
        }
        A2[j].x = lo ? sc * w_ih[16 * tg + m] : 0.0f;
        A2[j].y = lo ? sc * (b_ih[16 * tg + m] + b_hh[16 * tg + m]) : 0.0f;
    }
    v2f B2;
    B2.y = lo ? 1.0f : 0.0f;                       // ones row for bias

    const int len = lengths[base + m];
    int mx = len;
#pragma unroll
    for (int off = 16; off >= 1; off >>= 1)
        mx = max(mx, __shfl_xor(mx, off, 32));

    const float* xrow = x + (size_t)(base + m) * T_SZ;

    v8f c = {}, h = {};                            // this wave's hidden half

    // LDS exchange addressing (uint4 units):
    //   write slot:  [buf][wave p][lane]
    //   B-operand:   dwords 0..3 at r1, dwords 4..7 at r1+16 (contiguous!)
    const int r1 = lo ? lane : lane + 16;
    const int wr = p * 32 + lane;
    int bo = 0;                                    // buffer toggle: 0 / 64

    lds4[wr] = make_uint4(0u, 0u, 0u, 0u);         // h(init) = 0
    __syncthreads();

    for (int t = mx - 1; t >= 0; --t) {
        const float xt = xrow[t];
        __builtin_prefetch(xrow + t - 16, 0, 3);   // backward stream
        B2.x = lo ? xt : 0.0f;

        // full h as f16 B-matrix: two contiguous b128 LDS loads
        const uint4 la = lds4[bo + r1];
        const uint4 lb = lds4[bo + r1 + 16];
        v8u hbu;
        hbu[0] = la.x; hbu[1] = la.y; hbu[2] = la.z; hbu[3] = la.w;
        hbu[4] = lb.x; hbu[5] = lb.y; hbu[6] = lb.z; hbu[7] = lb.w;
        const v16h hB = __builtin_bit_cast(v16h, hbu);
        const v8f  z  = {};

        v8f d0 = __builtin_amdgcn_wmma_f32_16x16x4_f32(false, A2[0], false, B2, (short)0, z, false, false);
        d0 = __builtin_amdgcn_wmma_f32_16x16x32_f16(false, W[0], false, hB, (short)0, d0, false, false);
        v8f d1 = __builtin_amdgcn_wmma_f32_16x16x4_f32(false, A2[1], false, B2, (short)0, z, false, false);
        d1 = __builtin_amdgcn_wmma_f32_16x16x32_f16(false, W[1], false, hB, (short)0, d1, false, false);
        v8f d2 = __builtin_amdgcn_wmma_f32_16x16x4_f32(false, A2[2], false, B2, (short)0, z, false, false);
        d2 = __builtin_amdgcn_wmma_f32_16x16x32_f16(false, W[2], false, hB, (short)0, d2, false, false);
        v8f d3 = __builtin_amdgcn_wmma_f32_16x16x4_f32(false, A2[3], false, B2, (short)0, z, false, false);
        d3 = __builtin_amdgcn_wmma_f32_16x16x32_f16(false, W[3], false, hB, (short)0, d3, false, false);

        const bool upd = (t < len);                // packed-sequence mask
#pragma unroll
        for (int v = 0; v < 8; ++v) {
            const float iv = fmaf(0.5f, tanh_f(d0[v]), 0.5f);
            const float fv = fmaf(0.5f, tanh_f(d1[v]), 0.5f);
            const float gv = tanh_f(d2[v]);
            const float ov = fmaf(0.5f, tanh_f(d3[v]), 0.5f);
            const float cn = fmaf(fv, c[v], iv * gv);
            const float hn = ov * tanh_f(cn);
            c[v] = upd ? cn : c[v];
            h[v] = upd ? hn : h[v];
        }

        // pack own h tile to f16 and publish for next step (other buffer)
        uint4 pk;
        pk.x = __builtin_bit_cast(unsigned int, __builtin_amdgcn_cvt_pkrtz(h[0], h[1]));
        pk.y = __builtin_bit_cast(unsigned int, __builtin_amdgcn_cvt_pkrtz(h[2], h[3]));
        pk.z = __builtin_bit_cast(unsigned int, __builtin_amdgcn_cvt_pkrtz(h[4], h[5]));
        pk.w = __builtin_bit_cast(unsigned int, __builtin_amdgcn_cvt_pkrtz(h[6], h[7]));
        lds4[(bo ^ 64) + wr] = pk;
        __syncthreads();                           // single barrier per step
        bo ^= 64;
    }

    // --- write h_n (f32): wave p covers hidden 16p+8*ph .. +7 ---
    float* hw = h_ws + (size_t)(base + m) * H_SZ;
#pragma unroll
    for (int v = 0; v < 8; ++v)
        hw[16 * p + 8 * ph + v] = h[v];
}

// ---------------- MLP head: elu(h @ fc_w.T + fc_b) -> sigmoid(fc2) ----------
__global__ __launch_bounds__(256)
void head_kernel(const float* __restrict__ h_ws,
                 const float* __restrict__ fc_w, const float* __restrict__ fc_b,
                 const float* __restrict__ fc2_w, const float* __restrict__ fc2_b,
                 float* __restrict__ out)
{
    const int b = blockIdx.x * blockDim.x + threadIdx.x;
    if (b >= B_SZ) return;
    float h[H_SZ];
#pragma unroll
    for (int j = 0; j < H_SZ; ++j) h[j] = h_ws[(size_t)b * H_SZ + j];
    float acc = fc2_b[0];
    for (int k = 0; k < 64; ++k) {
        float s = fc_b[k];
#pragma unroll
        for (int j = 0; j < H_SZ; ++j) s = fmaf(fc_w[k * H_SZ + j], h[j], s);
        const float e = (s > 0.0f) ? s : (__expf(s) - 1.0f);   // ELU(alpha=1)
        acc = fmaf(fc2_w[k], e, acc);
    }
    out[b] = 1.0f / (1.0f + __expf(-acc));
}

extern "C" void kernel_launch(void* const* d_in, const int* in_sizes, int n_in,
                              void* d_out, int out_size, void* d_ws, size_t ws_size,
                              hipStream_t stream)
{
    const float* x     = (const float*)d_in[0];
    const int*   lens  = (const int*)  d_in[1];
    const float* w_ih  = (const float*)d_in[2];
    const float* w_hh  = (const float*)d_in[3];
    const float* b_ih  = (const float*)d_in[4];
    const float* b_hh  = (const float*)d_in[5];
    const float* fc_w  = (const float*)d_in[6];
    const float* fc_b  = (const float*)d_in[7];
    const float* fc2_w = (const float*)d_in[8];
    const float* fc2_b = (const float*)d_in[9];
    float* out  = (float*)d_out;
    float* h_ws = (float*)d_ws;        // 4096*32 f32 = 512 KB scratch

    // 256 blocks x (2 waves, 16 batches each) = 512 waves.
    lstm_bwd_kernel<<<256, 64, 0, stream>>>(x, lens, w_ih, w_hh, b_ih, b_hh, h_ws);
    head_kernel<<<B_SZ / 256, 256, 0, stream>>>(h_ws, fc_w, fc_b, fc2_w, fc2_b, out);
}
